// GroupedQueryAttention_51737176048321
// MI455X (gfx1250) — matrix operand order repaired and checked by reading, not compiled, not run on hardware
//
#include <hip/hip_runtime.h>
#include <hip/hip_bf16.h>

// ---------- CDNA5 WMMA vector types ----------
typedef __attribute__((ext_vector_type(16))) __bf16 v16bf;
typedef __attribute__((ext_vector_type(8)))  float  v8f;
typedef int v4i __attribute__((vector_size(16)));   // 16B staging unit

#define B_SZ 2
#define T_SZ 1024
#define DMODEL 3072
#define NH 32
#define NKV 8
#define HD 96
#define MTOT (B_SZ * T_SZ)   // 2048

// ---------- optional CDNA5 async global->LDS path (ASYNCcnt) ----------
// prototype (per hipcc diagnostics): (v4i* global_src, v4i __shared__* lds_dst, Ii off, Ii cpol)
#if defined(__HIP_DEVICE_COMPILE__) && defined(__gfx1250__) && \
    __has_builtin(__builtin_amdgcn_global_load_async_to_lds_b128)
#define HAVE_ASYNC_LDS 1
#if __has_builtin(__builtin_amdgcn_s_wait_asynccnt)
#define WAIT_ASYNC(n) __builtin_amdgcn_s_wait_asynccnt(n)
#else
#define WAIT_ASYNC(n) asm volatile("s_wait_asynccnt %0" ::"i"(n) : "memory")
#endif
#else
#define HAVE_ASYNC_LDS 0
#endif

// =====================================================================
// f32 -> bf16 conversion (grid-stride)
// =====================================================================
__global__ void cvt_bf16_kernel(const float* __restrict__ in, __bf16* __restrict__ out, long n) {
  long i = (long)blockIdx.x * blockDim.x + threadIdx.x;
  long st = (long)gridDim.x * blockDim.x;
  for (; i < n; i += st) out[i] = (__bf16)in[i];
}

// =====================================================================
// GEMM: C[M,N] (f32) = A[M,K] (bf16 row-major) * Bt[N,K]^T (bf16 row-major)
// Workgroup tile 128x128, 8 waves of 32x64, K-step 32, double-buffered LDS
// with async global->LDS copies when available.
// =====================================================================
__global__ __launch_bounds__(256) void gemm_bf16_kernel(
    const __bf16* __restrict__ A, const __bf16* __restrict__ Bt,
    float* __restrict__ C, int M, int N, int K) {
  // rows of 6 x v4i = 96B (48 bf16): 32 data cols + padding, 16B aligned
  __shared__ v4i AsV[2][128][6];
  __shared__ v4i BsV[2][128][6];

  const int tid  = threadIdx.x;
  const int lane = tid & 31;
  const int wave = tid >> 5;       // 0..7
  const int wr   = wave >> 1;      // 0..3 -> 32-row strip
  const int wc   = wave & 1;       // 0..1 -> 64-col strip
  const int lo   = lane & 15;
  const int hi   = lane >> 4;

  const int mBlock = blockIdx.y * 128;
  const int nBlock = blockIdx.x * 128;

  const int crow  = tid >> 1;         // 0..127 staging row
  const int ccol  = (tid & 1) * 16;   // bf16 element offset: 0 or 16
  const int ccolv = (tid & 1) * 2;    // v4i offset: 0 or 2

  v8f acc[2][4];
  v8f z8 = {0.f,0.f,0.f,0.f,0.f,0.f,0.f,0.f};
  for (int i = 0; i < 2; ++i) for (int j = 0; j < 4; ++j) acc[i][j] = z8;

  const int nk = K >> 5;   // K/32 steps

#if HAVE_ASYNC_LDS
  // ---- double-buffered pipeline: tile k+1 streams via DMA under tile-k WMMAs
  auto stage = [&](int k0, int buf) {
    const __bf16* ga = A  + (size_t)(mBlock + crow) * K + k0 + ccol;
    const __bf16* gb = Bt + (size_t)(nBlock + crow) * K + k0 + ccol;
    // IOFFSET applies to both global and LDS addresses (ISA 10.x async ops)
    __builtin_amdgcn_global_load_async_to_lds_b128(
        (v4i*)(void*)ga, &AsV[buf][crow][ccolv], 0, 0);
    __builtin_amdgcn_global_load_async_to_lds_b128(
        (v4i*)(void*)ga, &AsV[buf][crow][ccolv], 16, 0);
    __builtin_amdgcn_global_load_async_to_lds_b128(
        (v4i*)(void*)gb, &BsV[buf][crow][ccolv], 0, 0);
    __builtin_amdgcn_global_load_async_to_lds_b128(
        (v4i*)(void*)gb, &BsV[buf][crow][ccolv], 16, 0);
  };
  stage(0, 0);
  for (int kt = 0; kt < nk; ++kt) {
    const int buf = kt & 1;
    if (kt + 1 < nk) {
      stage((kt + 1) << 5, buf ^ 1);  // issue next tile (4 new async ops)
      WAIT_ASYNC(4);                  // own tile-k ops done (in-order per wave)
    } else {
      WAIT_ASYNC(0);
    }
    __syncthreads();                  // all waves' tile-k copies visible

    v16bf afrag[2], bfrag[4];
    for (int mi = 0; mi < 2; ++mi) {
      const __bf16* ap = (const __bf16*)&AsV[buf][wr * 32 + mi * 16 + lo][0];
      union { uint4 u[2]; v16bf v; } t;
      t.u[0] = *(const uint4*)(ap + hi * 8);
      t.u[1] = *(const uint4*)(ap + 16 + hi * 8);
      afrag[mi] = t.v;
    }
    for (int ni = 0; ni < 4; ++ni) {
      const __bf16* bp = (const __bf16*)&BsV[buf][wc * 64 + ni * 16 + lo][0];
      union { uint4 u[2]; v16bf v; } t;
      t.u[0] = *(const uint4*)(bp + hi * 8);
      t.u[1] = *(const uint4*)(bp + 16 + hi * 8);
      bfrag[ni] = t.v;
    }
    for (int mi = 0; mi < 2; ++mi)
      for (int ni = 0; ni < 4; ++ni)
        acc[mi][ni] = __builtin_amdgcn_wmma_f32_16x16x32_bf16(
            false, afrag[mi], false, bfrag[ni], (short)0, acc[mi][ni], false, false);
    __syncthreads();                  // reads done before buf is re-filled
  }
#else
  // ---- fallback: synchronous staging (proven path) ----
  for (int kt = 0; kt < nk; ++kt) {
    const int k0 = kt << 5;
    __syncthreads();
    const v4i* ga = (const v4i*)(A + (size_t)(mBlock + crow) * K + k0 + ccol);
    const v4i* gb = (const v4i*)(Bt + (size_t)(nBlock + crow) * K + k0 + ccol);
    v4i a0 = ga[0], a1 = ga[1];
    v4i b0 = gb[0], b1 = gb[1];
    if (k0 + 32 < K) {   // prefetch next K tile -> global_prefetch_b8
      __builtin_prefetch(A + (size_t)(mBlock + crow) * K + k0 + 32 + ccol, 0, 1);
      __builtin_prefetch(Bt + (size_t)(nBlock + crow) * K + k0 + 32 + ccol, 0, 1);
    }
    AsV[0][crow][ccolv]     = a0;
    AsV[0][crow][ccolv + 1] = a1;
    BsV[0][crow][ccolv]     = b0;
    BsV[0][crow][ccolv + 1] = b1;
    __syncthreads();

    v16bf afrag[2], bfrag[4];
    for (int mi = 0; mi < 2; ++mi) {
      const __bf16* ap = (const __bf16*)&AsV[0][wr * 32 + mi * 16 + lo][0];
      union { uint4 u[2]; v16bf v; } t;
      t.u[0] = *(const uint4*)(ap + hi * 8);
      t.u[1] = *(const uint4*)(ap + 16 + hi * 8);
      afrag[mi] = t.v;
    }
    for (int ni = 0; ni < 4; ++ni) {
      const __bf16* bp = (const __bf16*)&BsV[0][wc * 64 + ni * 16 + lo][0];
      union { uint4 u[2]; v16bf v; } t;
      t.u[0] = *(const uint4*)(bp + hi * 8);
      t.u[1] = *(const uint4*)(bp + 16 + hi * 8);
      bfrag[ni] = t.v;
    }
    for (int mi = 0; mi < 2; ++mi)
      for (int ni = 0; ni < 4; ++ni)
        acc[mi][ni] = __builtin_amdgcn_wmma_f32_16x16x32_bf16(
            false, afrag[mi], false, bfrag[ni], (short)0, acc[mi][ni], false, false);
  }
#endif

  // ---- store C (f32 16x16 layout: row = r + 8*hi, col = lo) ----
  for (int mi = 0; mi < 2; ++mi)
    for (int ni = 0; ni < 4; ++ni)
      for (int r = 0; r < 8; ++r) {
        int row = mBlock + wr * 32 + mi * 16 + r + 8 * hi;
        int col = nBlock + wc * 64 + ni * 16 + lo;
        C[(size_t)row * N + col] = acc[mi][ni][r];
      }
}

// =====================================================================
// RMSNorm + RoPE (+ layout shuffle) : one wave per (b,t,head) row of 96
//   q -> qb [b,h,t,96] bf16 ;  k -> kb [b,kv,t,96] bf16 ; v -> vtb [b,kv,96,1024]
// =====================================================================
__device__ inline float wave_sum32(float v) {
  v += __shfl_xor(v, 1, 32);  v += __shfl_xor(v, 2, 32);
  v += __shfl_xor(v, 4, 32);  v += __shfl_xor(v, 8, 32);
  v += __shfl_xor(v, 16, 32);
  return v;
}

__global__ __launch_bounds__(32) void normrope_kernel(
    const float* __restrict__ qf, const float* __restrict__ kf, const float* __restrict__ vf,
    const float* __restrict__ qw, const float* __restrict__ kw,
    const float* __restrict__ cosb, const float* __restrict__ sinb,
    __bf16* __restrict__ qb, __bf16* __restrict__ kb, __bf16* __restrict__ vtb) {
  const int QROWS = MTOT * NH;     // 65536
  const int KROWS = MTOT * NKV;    // 16384
  int row  = blockIdx.x;
  int lane = threadIdx.x;

  const float* src; const float* w; __bf16* dst; int t;
  bool is_v = false;
  if (row < QROWS) {
    int h = row & 31; t = (row >> 5) & 1023; int b = row >> 15;
    src = qf + ((size_t)(b * 1024 + t)) * (NH * HD) + h * HD;
    w = qw;
    dst = qb + ((size_t)((b * NH + h) * 1024 + t)) * HD;
  } else if (row < QROWS + KROWS) {
    int r2 = row - QROWS;
    int kv = r2 & 7; t = (r2 >> 3) & 1023; int b = r2 >> 13;
    src = kf + ((size_t)(b * 1024 + t)) * (NKV * HD) + kv * HD;
    w = kw;
    dst = kb + ((size_t)((b * NKV + kv) * 1024 + t)) * HD;
  } else {
    int r2 = row - QROWS - KROWS;
    int kv = r2 & 7; t = (r2 >> 3) & 1023; int b = r2 >> 13;
    src = vf + ((size_t)(b * 1024 + t)) * (NKV * HD) + kv * HD;
    dst = vtb + ((size_t)(b * NKV + kv)) * (HD * 1024);
    is_v = true; w = nullptr;
  }

  if (is_v) {   // transpose-store only: vtb[...][d][t]
    for (int d = lane; d < HD; d += 32)
      dst[(size_t)d * 1024 + t] = (__bf16)src[d];
    return;
  }

  // pairs: lane handles pair p0 = lane (always), p1 = lane+32 (lanes 0..15)
  int p0 = lane, p1 = lane + 32;
  float e0 = src[2 * p0], o0 = src[2 * p0 + 1];
  float e1 = 0.f, o1 = 0.f;
  if (lane < 16) { e1 = src[2 * p1]; o1 = src[2 * p1 + 1]; }
  float ss = e0 * e0 + o0 * o0 + e1 * e1 + o1 * o1;
  ss = wave_sum32(ss);
  float inv = rsqrtf(ss / (float)HD + 1e-6f);

  const float* cr = cosb + (size_t)t * HD;
  const float* sr = sinb + (size_t)t * HD;
  // pair p: out[2p]   = n_e*cos[2p]   - n_o*sin[2p]
  //         out[2p+1] = n_e*sin[2p+1] + n_o*cos[2p+1]
  {
    float ne = w[2 * p0] * e0 * inv, no = w[2 * p0 + 1] * o0 * inv;
    dst[2 * p0]     = (__bf16)(ne * cr[2 * p0]     - no * sr[2 * p0]);
    dst[2 * p0 + 1] = (__bf16)(ne * sr[2 * p0 + 1] + no * cr[2 * p0 + 1]);
  }
  if (lane < 16) {
    float ne = w[2 * p1] * e1 * inv, no = w[2 * p1 + 1] * o1 * inv;
    dst[2 * p1]     = (__bf16)(ne * cr[2 * p1]     - no * sr[2 * p1]);
    dst[2 * p1 + 1] = (__bf16)(ne * sr[2 * p1 + 1] + no * cr[2 * p1 + 1]);
  }
}

// =====================================================================
// Causal flash attention: one wave per (b, h, 16-query tile)
// =====================================================================
__global__ __launch_bounds__(32) void attn_kernel(
    const __bf16* __restrict__ qb, const __bf16* __restrict__ kb,
    const __bf16* __restrict__ vtb, __bf16* __restrict__ ab) {
  __shared__ __bf16 Ps[16][48];    // P tile round-trip (C-layout -> A-layout)

  int wid = blockIdx.x;
  int qt = wid & 63;
  int h  = (wid >> 6) & 31;
  int b  = wid >> 11;
  int kv = h >> 2;                 // 4:1 GQA
  int lane = threadIdx.x;
  int lo = lane & 15, hi = lane >> 4;
  int qbase = qt * 16;

  const __bf16* Qh = qb  + ((size_t)(b * NH + h)  * 1024) * HD;
  const __bf16* Kh = kb  + ((size_t)(b * NKV + kv) * 1024) * HD;
  const __bf16* Vh = vtb + ((size_t)(b * NKV + kv) * HD) * 1024;

  const float scale = 0.1020620726f;   // 1/sqrt(96)
  const v8f z8 = {0.f,0.f,0.f,0.f,0.f,0.f,0.f,0.f};

  // Q fragments (resident): A-matrix layout, 3 chunks of K=32 over d
  v16bf qfrag[3];
  {
    const __bf16* qp = Qh + (size_t)(qbase + lo) * HD;
    for (int kc = 0; kc < 3; ++kc) {
      union { uint4 u[2]; v16bf v; } tu;
      tu.u[0] = *(const uint4*)(qp + kc * 32 + hi * 8);
      tu.u[1] = *(const uint4*)(qp + kc * 32 + 16 + hi * 8);
      qfrag[kc] = tu.v;
    }
  }

  float m_r[8], l_r[8];
  for (int r = 0; r < 8; ++r) { m_r[r] = -3.0e38f; l_r[r] = 0.f; }
  v8f acc[6];
  for (int nt = 0; nt < 6; ++nt) acc[nt] = z8;

  for (int k0 = 0; k0 < qbase + 16; k0 += 32) {
    // ---- scores for two 16-key halves ----
    v8f sc[2];
    for (int hh = 0; hh < 2; ++hh) {
      int kt = k0 + hh * 16;
      v8f s = z8;
      if (kt < qbase + 16) {       // otherwise fully masked: leave zeros
        const __bf16* kp = Kh + (size_t)(kt + lo) * HD + hi * 8;
        for (int kc = 0; kc < 3; ++kc) {
          union { uint4 u[2]; v16bf v; } tu;
          tu.u[0] = *(const uint4*)(kp + kc * 32);
          tu.u[1] = *(const uint4*)(kp + kc * 32 + 16);
          s = __builtin_amdgcn_wmma_f32_16x16x32_bf16(
              false, qfrag[kc], false, tu.v, (short)0, s, false, false);
        }
      }
      for (int r = 0; r < 8; ++r) {
        int qi = qbase + r + 8 * hi;
        int key = kt + lo;
        s[r] = s[r] * scale + (key <= qi ? 0.f : -1e9f);
      }
      sc[hh] = s;
    }

    // ---- streaming softmax update (row stats across 16-lane halves) ----
    float alpha[8];
    for (int r = 0; r < 8; ++r) {
      float v = fmaxf(sc[0][r], sc[1][r]);
      v = fmaxf(v, __shfl_xor(v, 1, 32));
      v = fmaxf(v, __shfl_xor(v, 2, 32));
      v = fmaxf(v, __shfl_xor(v, 4, 32));
      v = fmaxf(v, __shfl_xor(v, 8, 32));
      float mn = fmaxf(m_r[r], v);
      float a  = __expf(m_r[r] - mn);
      float p0 = __expf(sc[0][r] - mn);
      float p1 = __expf(sc[1][r] - mn);
      float rs = p0 + p1;
      rs += __shfl_xor(rs, 1, 32);
      rs += __shfl_xor(rs, 2, 32);
      rs += __shfl_xor(rs, 4, 32);
      rs += __shfl_xor(rs, 8, 32);
      l_r[r] = l_r[r] * a + rs;
      m_r[r] = mn;
      alpha[r] = a;
      sc[0][r] = p0; sc[1][r] = p1;
    }
    for (int nt = 0; nt < 6; ++nt)
      for (int r = 0; r < 8; ++r) acc[nt][r] *= alpha[r];

    // ---- P (16x32) through LDS: C-layout stores, A-layout loads ----
    for (int r = 0; r < 8; ++r) {
      Ps[r + 8 * hi][lo]      = (__bf16)sc[0][r];
      Ps[r + 8 * hi][16 + lo] = (__bf16)sc[1][r];
    }
    __syncthreads();
    v16bf pfrag;
    {
      union { uint4 u[2]; v16bf v; } tu;
      tu.u[0] = *(const uint4*)&Ps[lo][hi * 8];
      tu.u[1] = *(const uint4*)&Ps[lo][16 + hi * 8];
      pfrag = tu.v;
    }

    // ---- O += P * V (V transposed: contiguous keys per dim) ----
    for (int nt = 0; nt < 6; ++nt) {
      const __bf16* vp = Vh + (size_t)(nt * 16 + lo) * 1024 + k0 + hi * 8;
      union { uint4 u[2]; v16bf v; } tu;
      tu.u[0] = *(const uint4*)vp;
      tu.u[1] = *(const uint4*)(vp + 16);
      acc[nt] = __builtin_amdgcn_wmma_f32_16x16x32_bf16(
          false, pfrag, false, tu.v, (short)0, acc[nt], false, false);
    }
    __syncthreads();
  }

  // ---- normalize + store bf16 into [b,t,h,96] for the O-projection ----
  for (int nt = 0; nt < 6; ++nt) {
    int dim = nt * 16 + lo;
    for (int r = 0; r < 8; ++r) {
      int row = qbase + r + 8 * hi;
      ab[((size_t)(b * 1024 + row) * NH + h) * HD + dim] = (__bf16)(acc[nt][r] / l_r[r]);
    }
  }
}

// =====================================================================
// Host launcher
// =====================================================================
extern "C" void kernel_launch(void* const* d_in, const int* in_sizes, int n_in,
                              void* d_out, int out_size, void* d_ws, size_t ws_size,
                              hipStream_t stream) {
  (void)in_sizes; (void)n_in; (void)out_size; (void)ws_size;

  const float* x    = (const float*)d_in[0];
  // d_in[1] = mask (causal; implemented analytically)
  const float* wq   = (const float*)d_in[2];
  const float* wk   = (const float*)d_in[3];
  const float* wv   = (const float*)d_in[4];
  const float* wo   = (const float*)d_in[5];
  const float* qnw  = (const float*)d_in[6];
  const float* knw  = (const float*)d_in[7];
  const float* cosb = (const float*)d_in[8];
  const float* sinb = (const float*)d_in[9];
  float* out = (float*)d_out;

  // ---- workspace carve (256B aligned slabs) ----
  char* base = (char*)d_ws;
  size_t off = 0;
  auto carve = [&](size_t bytes) -> void* {
    void* p = base + off;
    off = (off + bytes + 255) & ~(size_t)255;
    return p;
  };
  const long NE_X  = (long)MTOT * DMODEL;       // 6291456
  const long NE_WQ = (long)NH * HD * DMODEL;    // 9437184
  const long NE_WK = (long)NKV * HD * DMODEL;   // 2359296
  const long NE_KV = (long)B_SZ * NKV * T_SZ * HD; // 1572864

  __bf16* xb  = (__bf16*)carve(NE_X * 2);
  __bf16* wqb = (__bf16*)carve(NE_WQ * 2);
  __bf16* wkb = (__bf16*)carve(NE_WK * 2);
  __bf16* wvb = (__bf16*)carve(NE_WK * 2);
  __bf16* wob = (__bf16*)carve(NE_WQ * 2);
  float*  qf  = (float*)carve((long)MTOT * (NH * HD) * 4);
  float*  kf  = (float*)carve((long)MTOT * (NKV * HD) * 4);
  float*  vf  = (float*)carve((long)MTOT * (NKV * HD) * 4);
  __bf16* qbh = (__bf16*)carve(NE_X * 2);       // [b,h,t,96]
  __bf16* kbh = (__bf16*)carve(NE_KV * 2);      // [b,kv,t,96]
  __bf16* vtb = (__bf16*)carve(NE_KV * 2);      // [b,kv,96,1024]
  __bf16* ab  = (__bf16*)carve(NE_X * 2);       // [b,t,h,96]

  // 1) convert inputs/weights to bf16
  cvt_bf16_kernel<<<dim3(1024), dim3(256), 0, stream>>>(x,  xb,  NE_X);
  cvt_bf16_kernel<<<dim3(1024), dim3(256), 0, stream>>>(wq, wqb, NE_WQ);
  cvt_bf16_kernel<<<dim3(1024), dim3(256), 0, stream>>>(wk, wkb, NE_WK);
  cvt_bf16_kernel<<<dim3(1024), dim3(256), 0, stream>>>(wv, wvb, NE_WK);
  cvt_bf16_kernel<<<dim3(1024), dim3(256), 0, stream>>>(wo, wob, NE_WQ);

  // 2) QKV projections (C = X * W^T)
  gemm_bf16_kernel<<<dim3((NH * HD) / 128, MTOT / 128), 256, 0, stream>>>(
      xb, wqb, qf, MTOT, NH * HD, DMODEL);
  gemm_bf16_kernel<<<dim3((NKV * HD) / 128, MTOT / 128), 256, 0, stream>>>(
      xb, wkb, kf, MTOT, NKV * HD, DMODEL);
  gemm_bf16_kernel<<<dim3((NKV * HD) / 128, MTOT / 128), 256, 0, stream>>>(
      xb, wvb, vf, MTOT, NKV * HD, DMODEL);

  // 3) RMSNorm + RoPE + layout shuffles
  normrope_kernel<<<dim3(MTOT * (NH + 2 * NKV)), dim3(32), 0, stream>>>(
      qf, kf, vf, qnw, knw, cosb, sinb, qbh, kbh, vtb);

  // 4) causal flash attention
  attn_kernel<<<dim3(B_SZ * NH * (T_SZ / 16)), dim3(32), 0, stream>>>(qbh, kbh, vtb, ab);

  // 5) output projection straight into d_out (f32)
  gemm_bf16_kernel<<<dim3(DMODEL / 128, MTOT / 128), 256, 0, stream>>>(
      ab, wob, out, MTOT, DMODEL, NH * HD);
}
